// TGNAgent_58720792871839
// MI455X (gfx1250) — matrix operand order repaired and checked by reading, not compiled, not run on hardware
//
#include <hip/hip_runtime.h>

typedef __bf16 bf16;
typedef __attribute__((ext_vector_type(16))) __bf16 v16bf;
typedef __attribute__((ext_vector_type(8)))  float  v8f;
typedef __attribute__((ext_vector_type(4)))  float  f4;

static constexpr int BATCH = 2048;
static constexpr int AGENTS = 8;
static constexpr int NN = BATCH * AGENTS;   // 16384 nodes
static constexpr int DIN = 192;
static constexpr int HD  = 256;
static constexpr int MD  = 256;
static constexpr int GD  = 128;
static constexpr int NH  = 4;
static constexpr int NA  = 20;
static constexpr int EPG = AGENTS * (AGENTS - 1); // 56 edges per graph

// ---------------- WMMA fragment helpers (CDNA5 16x16x32 bf16) ----------------
// A: 16x32 bf16, row-major source. lane l: row = l&15, K = kb + (l>>4)*16 + i.
__device__ __forceinline__ v16bf load_a_frag(const bf16* __restrict__ A, int lda,
                                             int row0, int kb, int lane) {
  int m = lane & 15, g = lane >> 4;
  const bf16* p = A + (size_t)(row0 + m) * lda + kb + g * 16;
  v16bf a;
#pragma unroll
  for (int i = 0; i < 16; ++i) a[i] = p[i];       // K = kb + g*16 + i
  return a;
}
// B: 32x16 bf16, read from pre-transposed weights Bt[Ncols][K].
// lane l: col = l&15, K = kb + (l>>4)*16 + i  -> 16 contiguous bf16 per lane.
__device__ __forceinline__ v16bf load_b_frag(const bf16* __restrict__ Bt, int ldb,
                                             int n0, int kb, int lane) {
  int n = lane & 15, g = lane >> 4;
  const bf16* p = Bt + (size_t)(n0 + n) * ldb + kb + g * 16;
  v16bf b;
#pragma unroll
  for (int i = 0; i < 16; ++i) b[i] = p[i];
  return b;
}
__device__ __forceinline__ v8f wmma_bf16(v16bf a, v16bf b, v8f c) {
  return __builtin_amdgcn_wmma_f32_16x16x32_bf16(false, a, false, b, (short)0, c,
                                                 false, false);
}

// ---------------- f32 -> bf16 conversion kernels ----------------
__global__ void k_cvt(const float* __restrict__ in, bf16* __restrict__ out, int n) {
  for (int i = blockIdx.x * blockDim.x + threadIdx.x; i < n; i += blockDim.x * gridDim.x)
    out[i] = (bf16)in[i];
}
// in [rows][cols] f32 -> out [cols][rows] bf16 (weight transpose for B frags)
__global__ void k_cvt_t(const float* __restrict__ in, bf16* __restrict__ out,
                        int rows, int cols) {
  int n = rows * cols;
  for (int i = blockIdx.x * blockDim.x + threadIdx.x; i < n; i += blockDim.x * gridDim.x) {
    int r = i / cols, c = i % cols;
    out[(size_t)c * rows + r] = (bf16)in[i];
  }
}

// ---------------- edge_attr mean (deterministic 2-pass) ----------------
__global__ void k_ea_part(const float* __restrict__ ea, float* __restrict__ part, int E) {
  __shared__ float s[3][256];
  int tid = threadIdx.x;
  float a0 = 0.f, a1 = 0.f, a2 = 0.f;
  for (int i = blockIdx.x * 256 + tid; i < E; i += 256 * gridDim.x) {
    a0 += ea[(size_t)i * 3 + 0];
    a1 += ea[(size_t)i * 3 + 1];
    a2 += ea[(size_t)i * 3 + 2];
  }
  s[0][tid] = a0; s[1][tid] = a1; s[2][tid] = a2;
  __syncthreads();
  for (int st = 128; st > 0; st >>= 1) {
    if (tid < st) { s[0][tid] += s[0][tid + st]; s[1][tid] += s[1][tid + st]; s[2][tid] += s[2][tid + st]; }
    __syncthreads();
  }
  if (tid == 0) {
    part[blockIdx.x * 3 + 0] = s[0][0];
    part[blockIdx.x * 3 + 1] = s[1][0];
    part[blockIdx.x * 3 + 2] = s[2][0];
  }
}
__global__ void k_ea_fin(const float* __restrict__ part, float* __restrict__ eam,
                         int nblk, float invE) {
  int t = threadIdx.x;
  if (t < 3) {
    float s = 0.f;
    for (int i = 0; i < nblk; ++i) s += part[i * 3 + t];
    eam[t] = s * invE;
  }
}

// ---------------- MLP layer 1: x1 = LN(relu(in @ W1 + b1)) ----------------
__global__ __launch_bounds__(256) void k_mlp1(const bf16* __restrict__ A,
                                              const bf16* __restrict__ Bt,
                                              const float* __restrict__ bias,
                                              const float* __restrict__ gw,
                                              const float* __restrict__ gb,
                                              bf16* __restrict__ out) {
  int wave = threadIdx.x >> 5, lane = threadIdx.x & 31;
  int row0 = (blockIdx.x * 8 + wave) * 16;
  v8f zero = {0.f, 0.f, 0.f, 0.f, 0.f, 0.f, 0.f, 0.f};
  v8f acc[16];
#pragma unroll
  for (int t = 0; t < 16; ++t) acc[t] = zero;
  for (int k = 0; k < DIN; k += 32) {
    v16bf a = load_a_frag(A, DIN, row0, k, lane);
#pragma unroll
    for (int t = 0; t < 16; ++t)
      acc[t] = wmma_bf16(a, load_b_frag(Bt, DIN, t * 16, k, lane), acc[t]);
  }
  int c = lane & 15, grp = lane >> 4;
#pragma unroll
  for (int t = 0; t < 16; ++t) {
    float bv = bias[t * 16 + c];
#pragma unroll
    for (int r = 0; r < 8; ++r) acc[t][r] = fmaxf(acc[t][r] + bv, 0.f);
  }
  float mu[8], rs[8];
#pragma unroll
  for (int r = 0; r < 8; ++r) {
    float s1 = 0.f, s2 = 0.f;
#pragma unroll
    for (int t = 0; t < 16; ++t) { float v = acc[t][r]; s1 += v; s2 += v * v; }
#pragma unroll
    for (int off = 1; off < 16; off <<= 1) {
      s1 += __shfl_xor(s1, off, 32);
      s2 += __shfl_xor(s2, off, 32);
    }
    float m = s1 * (1.f / 256.f);
    mu[r] = m;
    rs[r] = rsqrtf(s2 * (1.f / 256.f) - m * m + 1e-5f);
  }
#pragma unroll
  for (int t = 0; t < 16; ++t) {
    int n = t * 16 + c;
    float g = gw[n], bb = gb[n];
#pragma unroll
    for (int r = 0; r < 8; ++r) {
      float y = (acc[t][r] - mu[r]) * rs[r] * g + bb;
      out[(size_t)(row0 + grp * 8 + r) * HD + n] = (bf16)y;
    }
  }
}

// ---- MLP layer 2 + fused SimpleConv(mean over graph): xm[B,256] ----
__global__ __launch_bounds__(256) void k_mlp2_mean(const bf16* __restrict__ A,
                                                   const bf16* __restrict__ Bt,
                                                   const float* __restrict__ bias,
                                                   const float* __restrict__ gw,
                                                   const float* __restrict__ gb,
                                                   bf16* __restrict__ xm) {
  int wave = threadIdx.x >> 5, lane = threadIdx.x & 31;
  int row0 = (blockIdx.x * 8 + wave) * 16;
  v8f zero = {0.f, 0.f, 0.f, 0.f, 0.f, 0.f, 0.f, 0.f};
  v8f acc[16];
#pragma unroll
  for (int t = 0; t < 16; ++t) acc[t] = zero;
  for (int k = 0; k < HD; k += 32) {
    v16bf a = load_a_frag(A, HD, row0, k, lane);
#pragma unroll
    for (int t = 0; t < 16; ++t)
      acc[t] = wmma_bf16(a, load_b_frag(Bt, HD, t * 16, k, lane), acc[t]);
  }
  int c = lane & 15, grp = lane >> 4;
#pragma unroll
  for (int t = 0; t < 16; ++t) {
    float bv = bias[t * 16 + c];
#pragma unroll
    for (int r = 0; r < 8; ++r) acc[t][r] = fmaxf(acc[t][r] + bv, 0.f);
  }
  float mu[8], rs[8];
#pragma unroll
  for (int r = 0; r < 8; ++r) {
    float s1 = 0.f, s2 = 0.f;
#pragma unroll
    for (int t = 0; t < 16; ++t) { float v = acc[t][r]; s1 += v; s2 += v * v; }
#pragma unroll
    for (int off = 1; off < 16; off <<= 1) {
      s1 += __shfl_xor(s1, off, 32);
      s2 += __shfl_xor(s2, off, 32);
    }
    float m = s1 * (1.f / 256.f);
    mu[r] = m;
    rs[r] = rsqrtf(s2 * (1.f / 256.f) - m * m + 1e-5f);
  }
  // rows row0+grp*8 .. +7 form exactly one graph (8 agents); mean over r is the
  // SimpleConv(mean, self_loop) output, identical for all nodes of the graph.
  int graph = (row0 >> 3) + grp;
#pragma unroll
  for (int t = 0; t < 16; ++t) {
    int n = t * 16 + c;
    float g = gw[n], bb = gb[n];
    float s = 0.f;
#pragma unroll
    for (int r = 0; r < 8; ++r) s += (acc[t][r] - mu[r]) * rs[r] * g + bb;
    xm[(size_t)graph * HD + n] = (bf16)(s * 0.125f);
  }
}

// ---------------- generic GEMM (+optional bias) -> f32 ----------------
template <int NT>
__global__ __launch_bounds__(256) void k_gemm_bias(const bf16* __restrict__ A,
                                                   const bf16* __restrict__ Bt,
                                                   const float* __restrict__ bias,
                                                   float* __restrict__ out,
                                                   int M, int K, int Ncols) {
  int wave = threadIdx.x >> 5, lane = threadIdx.x & 31;
  int row0 = (blockIdx.x * 8 + wave) * 16;
  if (row0 >= M) return;
  int n0 = blockIdx.y * (NT * 16);
  v8f zero = {0.f, 0.f, 0.f, 0.f, 0.f, 0.f, 0.f, 0.f};
  v8f acc[NT];
#pragma unroll
  for (int t = 0; t < NT; ++t) acc[t] = zero;
  for (int k = 0; k < K; k += 32) {
    v16bf a = load_a_frag(A, K, row0, k, lane);
#pragma unroll
    for (int t = 0; t < NT; ++t)
      acc[t] = wmma_bf16(a, load_b_frag(Bt, K, n0 + t * 16, k, lane), acc[t]);
  }
  int c = lane & 15, grp = lane >> 4;
#pragma unroll
  for (int t = 0; t < NT; ++t) {
    int n = n0 + t * 16 + c;
    float bv = bias ? bias[n] : 0.f;
#pragma unroll
    for (int r = 0; r < 8; ++r)
      out[(size_t)(row0 + grp * 8 + r) * Ncols + n] = acc[t][r] + bv;
  }
}

// ---- GRU: gh = h0 @ Whh^T (WMMA) fused with full GRU pointwise ----
__global__ __launch_bounds__(256) void k_gru(const bf16* __restrict__ Hbf,
                                             const bf16* __restrict__ Bt,
                                             const float* __restrict__ gi,
                                             const float* __restrict__ bhh,
                                             const float* __restrict__ h0,
                                             float* __restrict__ hout,
                                             bf16* __restrict__ hbf) {
  int wave = threadIdx.x >> 5, lane = threadIdx.x & 31;
  int row0 = (blockIdx.x * 8 + wave) * 16;
  int cc = blockIdx.y * 64;
  v8f zero = {0.f, 0.f, 0.f, 0.f, 0.f, 0.f, 0.f, 0.f};
  v8f aR[4], aZ[4], aN[4];
#pragma unroll
  for (int t = 0; t < 4; ++t) { aR[t] = zero; aZ[t] = zero; aN[t] = zero; }
  for (int k = 0; k < MD; k += 32) {
    v16bf a = load_a_frag(Hbf, MD, row0, k, lane);
#pragma unroll
    for (int t = 0; t < 4; ++t) {
      aR[t] = wmma_bf16(a, load_b_frag(Bt, MD, cc + t * 16, k, lane), aR[t]);
      aZ[t] = wmma_bf16(a, load_b_frag(Bt, MD, 256 + cc + t * 16, k, lane), aZ[t]);
      aN[t] = wmma_bf16(a, load_b_frag(Bt, MD, 512 + cc + t * 16, k, lane), aN[t]);
    }
  }
  int c = lane & 15, grp = lane >> 4;
  int graph = (row0 >> 3) + grp;  // rows in this half all belong to one graph
#pragma unroll
  for (int t = 0; t < 4; ++t) {
    int n = cc + t * 16 + c;
    float gir = gi[(size_t)graph * 768 + n];
    float giz = gi[(size_t)graph * 768 + 256 + n];
    float gin = gi[(size_t)graph * 768 + 512 + n];
    float br_ = bhh[n], bz_ = bhh[256 + n], bn_ = bhh[512 + n];
#pragma unroll
    for (int r = 0; r < 8; ++r) {
      int row = row0 + grp * 8 + r;
      float rr = 1.f / (1.f + __expf(-(gir + aR[t][r] + br_)));
      float zz = 1.f / (1.f + __expf(-(giz + aZ[t][r] + bz_)));
      float nn = tanhf(gin + rr * (aN[t][r] + bn_));
      float hp = h0[(size_t)row * MD + n];
      float hv = (1.f - zz) * nn + zz * hp;
      hout[(size_t)row * MD + n] = hv;
      hbf[(size_t)row * MD + n] = (bf16)hv;
    }
  }
}

// ---- GATv2 attention + head (one block per graph; wave = head) ----
__global__ __launch_bounds__(128) void k_gat(const float* __restrict__ xl,
                                             const float* __restrict__ xr,
                                             const float* __restrict__ hres,
                                             const float* __restrict__ eattr,
                                             const float* __restrict__ eam,
                                             const float* __restrict__ We,
                                             const float* __restrict__ att,
                                             const float* __restrict__ bg,
                                             const float* __restrict__ lng,
                                             const float* __restrict__ lnb,
                                             const float* __restrict__ Wout,
                                             const float* __restrict__ bout,
                                             float* __restrict__ q) {
  __shared__ float outh[NH][AGENTS][GD];
  __shared__ float zn[AGENTS][GD];
  int g = blockIdx.x;
  int h = threadIdx.x >> 5;
  int lane = threadIdx.x & 31;
  int d0 = lane * 4;

  f4 xlf[8], xrf[8];
#pragma unroll
  for (int j = 0; j < 8; ++j) {
    xlf[j] = *(const f4*)(xl + (size_t)(g * 8 + j) * (NH * GD) + h * GD + d0);
    xrf[j] = *(const f4*)(xr + (size_t)(g * 8 + j) * (NH * GD) + h * GD + d0);
  }
  f4 we0 = *(const f4*)(We + 0 * (NH * GD) + h * GD + d0);
  f4 we1 = *(const f4*)(We + 1 * (NH * GD) + h * GD + d0);
  f4 we2 = *(const f4*)(We + 2 * (NH * GD) + h * GD + d0);
  f4 av  = *(const f4*)(att + h * GD + d0);
  f4 ewself = we0 * eam[0] + we1 * eam[1] + we2 * eam[2];

  for (int i = 0; i < 8; ++i) {
    float logit[8];
#pragma unroll
    for (int j = 0; j < 8; ++j) {
      f4 ew;
      if (j == i) {
        ew = ewself;
      } else {
        int e = g * EPG + j * 7 + (i - (i > j ? 1 : 0));
        float a0 = eattr[(size_t)e * 3 + 0];
        float a1 = eattr[(size_t)e * 3 + 1];
        float a2 = eattr[(size_t)e * 3 + 2];
        ew = we0 * a0 + we1 * a1 + we2 * a2;
      }
      f4 m = xlf[j] + xrf[i] + ew;
      m.x = m.x > 0.f ? m.x : 0.2f * m.x;
      m.y = m.y > 0.f ? m.y : 0.2f * m.y;
      m.z = m.z > 0.f ? m.z : 0.2f * m.z;
      m.w = m.w > 0.f ? m.w : 0.2f * m.w;
      float p = m.x * av.x + m.y * av.y + m.z * av.z + m.w * av.w;
#pragma unroll
      for (int off = 16; off > 0; off >>= 1) p += __shfl_xor(p, off, 32);
      logit[j] = p;
    }
    float mx = logit[0];
#pragma unroll
    for (int j = 1; j < 8; ++j) mx = fmaxf(mx, logit[j]);
    float al[8];
    float den = 0.f;
#pragma unroll
    for (int j = 0; j < 8; ++j) { al[j] = __expf(logit[j] - mx); den += al[j]; }
    float inv = 1.f / (den + 1e-16f);
    f4 acc = xlf[0] * (al[0] * inv);
#pragma unroll
    for (int j = 1; j < 8; ++j) acc += xlf[j] * (al[j] * inv);
    *(f4*)(&outh[h][i][d0]) = acc;
  }
  __syncthreads();

  // head mean + residual + bg -> relu -> LN (16 threads per node)
  int t = threadIdx.x;
  int node = t >> 4, cc = t & 15;
  float zv[8];
  float s1 = 0.f, s2 = 0.f;
#pragma unroll
  for (int dd = 0; dd < 8; ++dd) {
    int d = cc * 8 + dd;
    float v = 0.25f * (outh[0][node][d] + outh[1][node][d] + outh[2][node][d] + outh[3][node][d]) +
              hres[(size_t)(g * 8 + node) * GD + d] + bg[d];
    v = fmaxf(v, 0.f);
    zv[dd] = v; s1 += v; s2 += v * v;
  }
#pragma unroll
  for (int off = 1; off < 16; off <<= 1) {
    s1 += __shfl_xor(s1, off, 32);
    s2 += __shfl_xor(s2, off, 32);
  }
  float mu = s1 * (1.f / GD);
  float rs = rsqrtf(s2 * (1.f / GD) - mu * mu + 1e-5f);
#pragma unroll
  for (int dd = 0; dd < 8; ++dd) {
    int d = cc * 8 + dd;
    zn[node][d] = (zv[dd] - mu) * rs * lng[d] + lnb[d];
  }
  __syncthreads();

  for (int idx = t; idx < AGENTS * NA; idx += 128) {
    int i2 = idx / NA, na = idx % NA;
    float acc = bout[na];
    for (int d = 0; d < GD; ++d) acc += zn[i2][d] * Wout[d * NA + na];
    q[(size_t)(g * 8 + i2) * NA + na] = acc;
  }
}

// ---------------- host-side launcher ----------------
extern "C" void kernel_launch(void* const* d_in, const int* in_sizes, int n_in,
                              void* d_out, int out_size, void* d_ws, size_t ws_size,
                              hipStream_t stream) {
  (void)n_in; (void)out_size; (void)ws_size;
  const float* inputs = (const float*)d_in[0];
  const float* h0     = (const float*)d_in[1];
  /* d_in[2] edge_index: topology is the fixed fully-connected graph; unused */
  const float* eattr  = (const float*)d_in[3];
  const float* W1 = (const float*)d_in[4],  *b1 = (const float*)d_in[5];
  const float* g1 = (const float*)d_in[6],  *be1 = (const float*)d_in[7];
  const float* W2 = (const float*)d_in[8],  *b2 = (const float*)d_in[9];
  const float* g2 = (const float*)d_in[10], *be2 = (const float*)d_in[11];
  const float* Wih = (const float*)d_in[12], *Whh = (const float*)d_in[13];
  const float* bih = (const float*)d_in[14], *bhh = (const float*)d_in[15];
  const float* Wl = (const float*)d_in[16], *bl = (const float*)d_in[17];
  const float* Wr = (const float*)d_in[18], *br = (const float*)d_in[19];
  const float* att = (const float*)d_in[20], *We = (const float*)d_in[21];
  const float* Wres = (const float*)d_in[22], *bg = (const float*)d_in[23];
  const float* lng = (const float*)d_in[24], *lnb = (const float*)d_in[25];
  const float* Wout = (const float*)d_in[26], *bout = (const float*)d_in[27];
  int E = in_sizes[3] / 3;

  float* q_out = (float*)d_out;
  float* h_out = (float*)d_out + (size_t)NN * NA;

  char* ws = (char*)d_ws;
  size_t off = 0;
  auto take = [&](size_t bytes) -> char* {
    char* p = ws + off;
    off += (bytes + 255) & ~(size_t)255;
    return p;
  };
  bf16* in_bf  = (bf16*)take((size_t)NN * DIN * 2);
  bf16* h0_bf  = (bf16*)take((size_t)NN * MD * 2);
  bf16* x1_bf  = (bf16*)take((size_t)NN * HD * 2);
  bf16* h_bf   = (bf16*)take((size_t)NN * MD * 2);
  bf16* xm_bf  = (bf16*)take((size_t)BATCH * HD * 2);
  bf16* Bt1    = (bf16*)take((size_t)HD * DIN * 2);
  bf16* Bt2    = (bf16*)take((size_t)HD * HD * 2);
  bf16* BtIh   = (bf16*)take((size_t)768 * HD * 2);
  bf16* BtHh   = (bf16*)take((size_t)768 * MD * 2);
  bf16* BtL    = (bf16*)take((size_t)(NH * GD) * MD * 2);
  bf16* BtR    = (bf16*)take((size_t)(NH * GD) * MD * 2);
  bf16* BtRes  = (bf16*)take((size_t)GD * MD * 2);
  float* gi    = (float*)take((size_t)BATCH * 768 * 4);
  float* xl    = (float*)take((size_t)NN * (NH * GD) * 4);
  float* xr    = (float*)take((size_t)NN * (NH * GD) * 4);
  float* hres  = (float*)take((size_t)NN * GD * 4);
  float* eapart = (float*)take((size_t)256 * 3 * 4);
  float* eam   = (float*)take((size_t)3 * 4);

  // convert activations + weights to bf16 (transpose weights used as x@W)
  k_cvt<<<512, 256, 0, stream>>>(inputs, in_bf, NN * DIN);
  k_cvt<<<512, 256, 0, stream>>>(h0, h0_bf, NN * MD);
  k_cvt_t<<<64, 256, 0, stream>>>(W1, Bt1, DIN, HD);
  k_cvt_t<<<64, 256, 0, stream>>>(W2, Bt2, HD, HD);
  k_cvt<<<128, 256, 0, stream>>>(Wih, BtIh, 768 * HD);   // (Wih^T)^T = Wih
  k_cvt<<<128, 256, 0, stream>>>(Whh, BtHh, 768 * MD);
  k_cvt_t<<<128, 256, 0, stream>>>(Wl, BtL, MD, NH * GD);
  k_cvt_t<<<128, 256, 0, stream>>>(Wr, BtR, MD, NH * GD);
  k_cvt_t<<<64, 256, 0, stream>>>(Wres, BtRes, MD, GD);

  // mean edge attribute (for self-loop fill)
  k_ea_part<<<256, 256, 0, stream>>>(eattr, eapart, E);
  k_ea_fin<<<1, 32, 0, stream>>>(eapart, eam, 256, 1.0f / (float)E);

  // base MLP (WMMA) + fused graph-mean conv
  k_mlp1<<<NN / 128, 256, 0, stream>>>(in_bf, Bt1, b1, g1, be1, x1_bf);
  k_mlp2_mean<<<NN / 128, 256, 0, stream>>>(x1_bf, Bt2, b2, g2, be2, xm_bf);

  // gi = xm @ Wih^T + bih  (per-graph: only 2048 rows)
  k_gemm_bias<16><<<dim3(BATCH / 128, 3), 256, 0, stream>>>(xm_bf, BtIh, bih, gi,
                                                            BATCH, HD, 768);
  // gh GEMM fused with GRU pointwise -> h (f32 to d_out) + h_bf
  k_gru<<<dim3(NN / 128, 4), 256, 0, stream>>>(h0_bf, BtHh, gi, bhh, h0, h_out, h_bf);

  // GAT projections
  k_gemm_bias<16><<<dim3(NN / 128, 2), 256, 0, stream>>>(h_bf, BtL, bl, xl, NN, MD, NH * GD);
  k_gemm_bias<16><<<dim3(NN / 128, 2), 256, 0, stream>>>(h_bf, BtR, br, xr, NN, MD, NH * GD);
  k_gemm_bias<8><<<dim3(NN / 128, 1), 256, 0, stream>>>(h_bf, BtRes, nullptr, hres, NN, MD, GD);

  // attention + head -> q
  k_gat<<<BATCH, 128, 0, stream>>>(xl, xr, hres, eattr, eam, We, att, bg, lng, lnb,
                                   Wout, bout, q_out);
}
